// MultiHeadAttention_11742440587388
// MI455X (gfx1250) — matrix operand order repaired and checked by reading, not compiled, not run on hardware
//
#include <hip/hip_runtime.h>

typedef __bf16 bf16;
typedef bf16  v16bf __attribute__((ext_vector_type(16)));
typedef bf16  v8bf  __attribute__((ext_vector_type(8)));
typedef float v8f   __attribute__((ext_vector_type(8)));

#define S_LEN 2048
#define D_DIM 2048
#define NH    16
#define HD    128
#define BATCH 2

static __device__ __forceinline__ v16bf cat8(v8bf lo, v8bf hi) {
  return __builtin_shufflevector(lo, hi, 0,1,2,3,4,5,6,7,8,9,10,11,12,13,14,15);
}

static __device__ __forceinline__ v8f wmma_bf16(v16bf a, v16bf b, v8f c) {
  // D = A(16x32 bf16) x B(32x16 bf16) + C(16x16 f32)
  return __builtin_amdgcn_wmma_f32_16x16x32_bf16(false, a, false, b, (short)0, c,
                                                 false, false);
}

// A-fragment (16x32 bf16): lane<16 -> row=lane, K {0..7,16..23}; lane>=16 -> K {8..15,24..31}
static __device__ __forceinline__ v16bf load_a_bf16(const bf16* base, int ld, int lane) {
  int r  = lane & 15;
  int kb = (lane >> 4) << 3;
  const bf16* p = base + (size_t)r * ld + kb;
  v8bf lo = *(const v8bf*)p;
  v8bf hi = *(const v8bf*)(p + 16);
  return cat8(lo, hi);
}

// Same A-fragment pattern, but from an LDS tile with fixed ld=32 (ds_load_b128).
static __device__ __forceinline__ v16bf load_a_lds(const bf16* base, int lane) {
  int r  = lane & 15;
  int kb = (lane >> 4) << 3;
  const bf16* p = base + r * 32 + kb;
  v8bf lo = *(const v8bf*)p;
  v8bf hi = *(const v8bf*)(p + 16);
  return cat8(lo, hi);
}

// B-fragment (32x16): column n = lane&15 taken from row n of a row-major (N x K) matrix;
// lane<16 holds K {0..7,16..23}, lane>=16 holds K {8..15,24..31} (matches A-fragment).
static __device__ __forceinline__ v16bf load_b_bf16(const bf16* base, int ld, int lane) {
  int n  = lane & 15;
  int kb = (lane >> 4) << 3;
  const bf16* p = base + (size_t)n * ld + kb;
  return cat8(*(const v8bf*)p, *(const v8bf*)(p + 16));
}

// Async copy of one 16-byte chunk per lane into LDS (ASYNCcnt-tracked DMA path).
static __device__ __forceinline__ void async_copy16(const bf16* g, const bf16* l) {
  unsigned lds = (unsigned)(size_t)l;              // low 32 bits = LDS offset
  unsigned long long ga = (unsigned long long)(size_t)g;
  asm volatile("global_load_async_to_lds_b128 %0, %1, off"
               :: "v"(lds), "v"(ga) : "memory");
}
static __device__ __forceinline__ void wait_async0() {
  asm volatile("s_wait_asynccnt 0x0" ::: "memory");
}

// ---------------------------------------------------------------------------
// Kernel 0: streaming f32 -> bf16 conversion (one pass; pays ~bandwidth once
// so GEMM hot loops read half-width operands with zero VALU converts).
// ---------------------------------------------------------------------------
__global__ __launch_bounds__(256) void cvt_f32_bf16_kernel(
    const float* __restrict__ in, bf16* __restrict__ out, int n8) {
  int i = blockIdx.x * 256 + threadIdx.x;
  if (i < n8) {
    v8f v = *(const v8f*)(in + (size_t)i * 8);
    v8bf o;
#pragma unroll
    for (int j = 0; j < 8; ++j) o[j] = (bf16)v[j];
    *(v8bf*)(out + (size_t)i * 8) = o;
  }
}

// ---------------------------------------------------------------------------
// Kernel 1: fused QKV projection.  qkv = x @ wqkv_w^T + b, scattered to
// q[B,H,S,hd] (bf16), k[B,H,S,hd] (bf16), v^T[B,H,hd,S] (bf16).
// grid (24, 64), block 256 = 8 waves; wave tile 32M x 64N (block 64M x 256N).
// A-panel (64x32) double-buffered in LDS via global_load_async_to_lds_b128.
// ---------------------------------------------------------------------------
__global__ __launch_bounds__(256) void qkv_gemm_kernel(
    const bf16* __restrict__ xb, const bf16* __restrict__ wb,
    const float* __restrict__ bias,
    bf16* __restrict__ qbuf, bf16* __restrict__ kbuf, bf16* __restrict__ vbufT) {
  __shared__ bf16 apanel[2][64 * 32];

  int tid  = threadIdx.x;
  int lane = tid & 31;
  int wid  = tid >> 5;
  int wm = wid & 1, wn = wid >> 1;
  int mblk = blockIdx.y * 64;
  int n0   = blockIdx.x * 256 + wn * 64;

  const bf16* abase = xb + (size_t)mblk * D_DIM;
  int arow = tid >> 2;          // 0..63: one row per 4 threads
  int ac8  = (tid & 3) << 3;    // 8-element (16B) chunk within the 32-wide row

  async_copy16(abase + (size_t)arow * D_DIM + ac8, &apanel[0][arow * 32 + ac8]);

  v8f acc[2][4] = {};
  for (int k0 = 0; k0 < D_DIM; k0 += 32) {
    wait_async0();          // this wave's panel chunk landed in LDS
    __syncthreads();        // every wave's chunk landed; prev readers drained
    int cur = (k0 >> 5) & 1;
    if (k0 + 32 < D_DIM)
      async_copy16(abase + (size_t)arow * D_DIM + (k0 + 32) + ac8,
                   &apanel[cur ^ 1][arow * 32 + ac8]);

    const bf16* ap = apanel[cur];
    v16bf a0 = load_a_lds(ap + (wm * 32 + 0) * 32, lane);
    v16bf a1 = load_a_lds(ap + (wm * 32 + 16) * 32, lane);
#pragma unroll
    for (int t = 0; t < 4; ++t) {
      const bf16* bp = wb + (size_t)(n0 + t * 16) * D_DIM + k0;
      __builtin_prefetch(bp + 32, 0, 3);   // next K-step of this weight panel
      v16bf b = load_b_bf16(bp, D_DIM, lane);
      acc[0][t] = wmma_bf16(a0, b, acc[0][t]);
      acc[1][t] = wmma_bf16(a1, b, acc[1][t]);
    }
  }

  int col = lane & 15, rb = (lane >> 4) << 3;
#pragma unroll
  for (int t = 0; t < 4; ++t) {
    int n     = n0 + t * 16 + col;
    int which = n >> 11;         // 0=q, 1=k, 2=v  (uniform over the 16-col tile)
    int d     = n & (D_DIM - 1);
    int h     = d >> 7;
    int di    = d & (HD - 1);
    float bv  = bias[n];
#pragma unroll
    for (int mi = 0; mi < 2; ++mi)
#pragma unroll
      for (int r = 0; r < 8; ++r) {
        int m = mblk + wm * 32 + mi * 16 + rb + r;
        int bidx = m >> 11;
        int s    = m & (S_LEN - 1);
        int bh   = bidx * NH + h;
        float v  = acc[mi][t][r] + bv;
        if (which == 0)
          qbuf[((size_t)bh * S_LEN + s) * HD + di] = (bf16)v;
        else if (which == 1)
          kbuf[((size_t)bh * S_LEN + s) * HD + di] = (bf16)v;
        else
          vbufT[((size_t)bh * HD + di) * S_LEN + s] = (bf16)v;
      }
  }
}

// ---------------------------------------------------------------------------
// Kernel 2: causal flash attention, one wave per (b,h, 16-query tile).
// grid = B*H*(S/16) = 4096 blocks of 32 threads.
// ---------------------------------------------------------------------------
__global__ __launch_bounds__(32) void attn_kernel(
    const bf16* __restrict__ qbuf, const bf16* __restrict__ kbuf,
    const bf16* __restrict__ vbufT, bf16* __restrict__ ctx) {
  __shared__ bf16 ptile[16 * 32];

  int lane = threadIdx.x & 31;
  int qi = blockIdx.x & ((S_LEN / 16) - 1);
  int bh = blockIdx.x >> 7;
  int s0 = qi * 16;
  const float scale = 0.08838834764831845f;  // 1/sqrt(128)

  v16bf qa[4];
#pragma unroll
  for (int kt = 0; kt < 4; ++kt)
    qa[kt] = load_a_bf16(qbuf + ((size_t)bh * S_LEN + s0) * HD + kt * 32, HD, lane);

  v8f acc[8] = {};
  float m_i[8], l_i[8];
#pragma unroll
  for (int r = 0; r < 8; ++r) { m_i[r] = -1e30f; l_i[r] = 0.0f; }

  int col = lane & 15, rb = (lane >> 4) << 3;

  for (int kk = 0; kk < s0 + 16; kk += 32) {
    // scores for keys kk..kk+15 and kk+16..kk+31
    v8f c0 = {}, c1 = {};
#pragma unroll
    for (int kt = 0; kt < 4; ++kt) {
      v16bf b0 = load_b_bf16(kbuf + ((size_t)bh * S_LEN + kk) * HD + kt * 32, HD, lane);
      c0 = wmma_bf16(qa[kt], b0, c0);
    }
#pragma unroll
    for (int kt = 0; kt < 4; ++kt) {
      v16bf b1 = load_b_bf16(kbuf + ((size_t)bh * S_LEN + kk + 16) * HD + kt * 32, HD, lane);
      c1 = wmma_bf16(qa[kt], b1, c1);
    }

    float corr[8];
#pragma unroll
    for (int r = 0; r < 8; ++r) {
      int qpos = s0 + rb + r;
      float v0 = (kk + col      <= qpos) ? c0[r] * scale : -1e30f;
      float v1 = (kk + 16 + col <= qpos) ? c1[r] * scale : -1e30f;
      // row max across the 16-lane group (closed under xor 1,2,4,8)
      float t = fmaxf(v0, v1);
      t = fmaxf(t, __shfl_xor(t, 1, 32));
      t = fmaxf(t, __shfl_xor(t, 2, 32));
      t = fmaxf(t, __shfl_xor(t, 4, 32));
      t = fmaxf(t, __shfl_xor(t, 8, 32));
      float mn = fmaxf(m_i[r], t);
      float cr = __expf(m_i[r] - mn);
      m_i[r] = mn;
      v0 = __expf(v0 - mn);
      v1 = __expf(v1 - mn);
      float rs = v0 + v1;
      rs += __shfl_xor(rs, 1, 32);
      rs += __shfl_xor(rs, 2, 32);
      rs += __shfl_xor(rs, 4, 32);
      rs += __shfl_xor(rs, 8, 32);
      l_i[r] = l_i[r] * cr + rs;
      corr[r] = cr;
      // stage P (C-layout) into LDS as bf16, row-major 16x32
      ptile[(rb + r) * 32 + col]      = (bf16)v0;
      ptile[(rb + r) * 32 + 16 + col] = (bf16)v1;
    }
#pragma unroll
    for (int nt = 0; nt < 8; ++nt)
#pragma unroll
      for (int r = 0; r < 8; ++r) acc[nt][r] *= corr[r];

    __syncthreads();
    {
      // reload P in A-fragment layout and accumulate P @ V
      v16bf pa = load_a_lds(ptile, lane);
#pragma unroll
      for (int nt = 0; nt < 8; ++nt) {
        v16bf bv = load_b_bf16(vbufT + ((size_t)bh * HD + nt * 16) * S_LEN + kk,
                               S_LEN, lane);
        acc[nt] = wmma_bf16(pa, bv, acc[nt]);
      }
    }
    __syncthreads();
  }

  int b = bh >> 4, h = bh & (NH - 1);
#pragma unroll
  for (int r = 0; r < 8; ++r) l_i[r] = 1.0f / l_i[r];
#pragma unroll
  for (int nt = 0; nt < 8; ++nt)
#pragma unroll
    for (int r = 0; r < 8; ++r) {
      int s = s0 + rb + r;
      ctx[((size_t)b * S_LEN + s) * D_DIM + h * HD + nt * 16 + col] =
          (bf16)(acc[nt][r] * l_i[r]);
    }
}

// ---------------------------------------------------------------------------
// Kernel 3: output projection.  out = ctx @ out_w^T + out_b  (f32 result)
// grid (8, 64), block 256; wave tile 32M x 64N; async LDS A-panel staging.
// ---------------------------------------------------------------------------
__global__ __launch_bounds__(256) void out_gemm_kernel(
    const bf16* __restrict__ ctx, const bf16* __restrict__ wb,
    const float* __restrict__ bias, float* __restrict__ out) {
  __shared__ bf16 apanel[2][64 * 32];

  int tid  = threadIdx.x;
  int lane = tid & 31;
  int wid  = tid >> 5;
  int wm = wid & 1, wn = wid >> 1;
  int mblk = blockIdx.y * 64;
  int n0   = blockIdx.x * 256 + wn * 64;

  const bf16* abase = ctx + (size_t)mblk * D_DIM;
  int arow = tid >> 2;
  int ac8  = (tid & 3) << 3;

  async_copy16(abase + (size_t)arow * D_DIM + ac8, &apanel[0][arow * 32 + ac8]);

  v8f acc[2][4] = {};
  for (int k0 = 0; k0 < D_DIM; k0 += 32) {
    wait_async0();
    __syncthreads();
    int cur = (k0 >> 5) & 1;
    if (k0 + 32 < D_DIM)
      async_copy16(abase + (size_t)arow * D_DIM + (k0 + 32) + ac8,
                   &apanel[cur ^ 1][arow * 32 + ac8]);

    const bf16* ap = apanel[cur];
    v16bf a0 = load_a_lds(ap + (wm * 32 + 0) * 32, lane);
    v16bf a1 = load_a_lds(ap + (wm * 32 + 16) * 32, lane);
#pragma unroll
    for (int t = 0; t < 4; ++t) {
      const bf16* bp = wb + (size_t)(n0 + t * 16) * D_DIM + k0;
      __builtin_prefetch(bp + 32, 0, 3);
      v16bf b = load_b_bf16(bp, D_DIM, lane);
      acc[0][t] = wmma_bf16(a0, b, acc[0][t]);
      acc[1][t] = wmma_bf16(a1, b, acc[1][t]);
    }
  }

  int col = lane & 15, rb = (lane >> 4) << 3;
#pragma unroll
  for (int t = 0; t < 4; ++t) {
    int n = n0 + t * 16 + col;
    float bv = bias[n];
#pragma unroll
    for (int mi = 0; mi < 2; ++mi)
#pragma unroll
      for (int r = 0; r < 8; ++r)
        out[(size_t)(mblk + wm * 32 + mi * 16 + rb + r) * D_DIM + n] =
            acc[mi][t][r] + bv;
  }
}

// ---------------------------------------------------------------------------
extern "C" void kernel_launch(void* const* d_in, const int* in_sizes, int n_in,
                              void* d_out, int out_size, void* d_ws, size_t ws_size,
                              hipStream_t stream) {
  const float* x      = (const float*)d_in[0];
  // d_in[1] is the boolean causal mask; applied analytically in attn_kernel.
  const float* wqkv_w = (const float*)d_in[2];
  const float* wqkv_b = (const float*)d_in[3];
  const float* out_w  = (const float*)d_in[4];
  const float* out_b  = (const float*)d_in[5];

  const size_t n_x    = (size_t)BATCH * S_LEN * D_DIM;  //  8,388,608
  const size_t n_wqkv = (size_t)3 * D_DIM * D_DIM;      // 12,582,912
  const size_t n_outw = (size_t)D_DIM * D_DIM;          //  4,194,304
  const size_t per    = (size_t)BATCH * NH * S_LEN * HD;//  8,388,608

  // Workspace layout (bf16 elements), with deterministic stream-ordered reuse:
  //   [0, n_wqkv)        : wqkv_w bf16  -> later reused for out_w bf16
  //   [n_wqkv, +n_x)     : x bf16       -> later reused for ctx
  //   then qbuf / kbuf / vbufT.   Total ~92 MB.
  bf16* wqkvb = (bf16*)d_ws;
  bf16* outwb = wqkvb;                 // alias: wqkv_w dead after GEMM1
  bf16* xb    = wqkvb + n_wqkv;
  bf16* ctx   = xb;                    // alias: x dead after GEMM1
  bf16* qbuf  = xb + n_x;
  bf16* kbuf  = qbuf + per;
  bf16* vbufT = kbuf + per;

  cvt_f32_bf16_kernel<<<dim3((unsigned)(n_x / 8 / 256)), 256, 0, stream>>>(
      x, xb, (int)(n_x / 8));
  cvt_f32_bf16_kernel<<<dim3((unsigned)(n_wqkv / 8 / 256)), 256, 0, stream>>>(
      wqkv_w, wqkvb, (int)(n_wqkv / 8));

  qkv_gemm_kernel<<<dim3(24, 64), 256, 0, stream>>>(xb, wqkvb, wqkv_b,
                                                    qbuf, kbuf, vbufT);
  attn_kernel<<<dim3(BATCH * NH * (S_LEN / 16)), 32, 0, stream>>>(qbuf, kbuf,
                                                                  vbufT, ctx);

  cvt_f32_bf16_kernel<<<dim3((unsigned)(n_outw / 8 / 256)), 256, 0, stream>>>(
      out_w, outwb, (int)(n_outw / 8));
  out_gemm_kernel<<<dim3(8, 64), 256, 0, stream>>>(ctx, outwb, out_b,
                                                   (float*)d_out);
}
